// seg_head_Deform_62311385531082
// MI455X (gfx1250) — matrix operand order repaired and checked by reading, not compiled, not run on hardware
//
#include <hip/hip_runtime.h>
#include <hip/hip_bf16.h>

#define B_    4
#define S_    5440
#define MTOT  (B_ * S_)     // 21760, multiple of 32
#define HID   256

typedef _Float16 half_t;
typedef __attribute__((ext_vector_type(16))) _Float16 v16h;
typedef __attribute__((ext_vector_type(8)))  float    v8f;

// ---------------------------------------------------------------------------
// Optional gfx1250 async global->LDS staging (ASYNCcnt-tracked)
// ---------------------------------------------------------------------------
#if defined(__has_builtin)
#  if __has_builtin(__builtin_amdgcn_global_load_async_to_lds_b128)
#    define USE_ASYNC_LDS 1
#  endif
#endif
#ifndef USE_ASYNC_LDS
#  define USE_ASYNC_LDS 0
#endif

#if USE_ASYNC_LDS
typedef int v4i_ __attribute__((vector_size(16)));
typedef __attribute__((address_space(1))) v4i_* g_v4i_ptr;
typedef __attribute__((address_space(3))) v4i_* l_v4i_ptr;
#endif

// stage a contiguous 8KB tile (16 rows x 256 halves) from global to LDS
__device__ __forceinline__ void stage_tile(const half_t* __restrict__ gsrc,
                                           half_t* __restrict__ lds_dst, int tid) {
#if USE_ASYNC_LDS
  g_v4i_ptr g = (g_v4i_ptr)(uintptr_t)gsrc;
  l_v4i_ptr l = (l_v4i_ptr)(unsigned)(uintptr_t)lds_dst;
#pragma unroll
  for (int i = 0; i < 2; ++i) {
    int idx = tid + i * 256;  // 512 x 16B chunks
    __builtin_amdgcn_global_load_async_to_lds_b128(g + idx, l + idx, 0, 0);
  }
#else
  for (int i = tid; i < 16 * 256; i += 256) lds_dst[i] = gsrc[i];
#endif
}

__device__ __forceinline__ void stage_wait() {
#if USE_ASYNC_LDS
#  if defined(__has_builtin) && __has_builtin(__builtin_amdgcn_s_wait_asynccnt)
  __builtin_amdgcn_s_wait_asynccnt(0);
#  else
  asm volatile("s_wait_asynccnt 0x0" ::: "memory");
#  endif
#endif
  __syncthreads();
}

// ---------------------------------------------------------------------------
// WMMA fragment helpers (CDNA5 16x16x32 f16 layouts, wave32)
// A (16x32, MxK): lanes 0-15 -> M=lane, halves 0..7 = K[k0..k0+7], 8..15 = K[k0+16..k0+23]
//                 lanes 16-31 -> halves 0..7 = K[k0+8..15], 8..15 = K[k0+24..31]
// B (32x16, KxN) for X@W^T with W row-major (N,K): lane n = lane&15,
//                 same K order as A: halves 0..7 = W[n][k0+kb+i], 8..15 = W[n][k0+16+kb+i], kb = lane>=16?8:0
// D (16x16 f32): elem r -> row m0 + r + (lane>=16?8:0), col n0 + (lane&15)
// ---------------------------------------------------------------------------
__device__ __forceinline__ v16h load_a_frag(const half_t* A, int lda, int m0, int k0, int lane) {
  int m  = m0 + (lane & 15);
  int kb = k0 + ((lane & 16) ? 8 : 0);
  const half_t* p = A + (size_t)m * lda + kb;
  v16h a;
#pragma unroll
  for (int i = 0; i < 8; ++i) a[i] = p[i];
#pragma unroll
  for (int i = 0; i < 8; ++i) a[8 + i] = p[16 + i];
  return a;
}

__device__ __forceinline__ v16h load_b_frag(const half_t* W, int ldw, int n0, int k0, int lane) {
  int n  = n0 + (lane & 15);
  int kb = k0 + ((lane & 16) ? 8 : 0);
  const half_t* p = W + (size_t)n * ldw + kb;
  v16h b;
#pragma unroll
  for (int i = 0; i < 8; ++i) b[i] = p[i];
#pragma unroll
  for (int i = 0; i < 8; ++i) b[8 + i] = p[16 + i];
  return b;
}

__device__ __forceinline__ v8f wmma_f16(v16h a, v16h b, v8f c) {
  return __builtin_amdgcn_wmma_f32_16x16x32_f16(false, a, false, b, (short)0, c, false, false);
}

// ---------------------------------------------------------------------------
// Generic GEMM: C[M,N] = A(f16 MxK) @ W(f16 NxK)^T + bias
// grid = (M/32, N/64), block = 32 (one wave computes 32x64: 8 WMMA per 6 loads)
// ---------------------------------------------------------------------------
__global__ void wmma_gemm_bias(const half_t* __restrict__ A, const half_t* __restrict__ W,
                               const float* __restrict__ bias, float* __restrict__ C,
                               int N, int K) {
  int lane = threadIdx.x;
  int m0 = blockIdx.x * 32;
  int n0 = blockIdx.y * 64;
  v8f acc[2][4] = {};
  for (int k0 = 0; k0 < K; k0 += 32) {
    v16h a0 = load_a_frag(A, K, m0,      k0, lane);
    v16h a1 = load_a_frag(A, K, m0 + 16, k0, lane);
#pragma unroll
    for (int j = 0; j < 4; ++j) {
      v16h b = load_b_frag(W, K, n0 + j * 16, k0, lane);
      acc[0][j] = wmma_f16(a0, b, acc[0][j]);
      acc[1][j] = wmma_f16(a1, b, acc[1][j]);
    }
  }
  int nc = lane & 15;
  int mrb = (lane & 16) ? 8 : 0;
#pragma unroll
  for (int mi = 0; mi < 2; ++mi) {
    int mr = m0 + mi * 16 + mrb;
#pragma unroll
    for (int j = 0; j < 4; ++j) {
      int n = n0 + j * 16 + nc;
      float bz = bias ? bias[n] : 0.0f;
#pragma unroll
      for (int r = 0; r < 8; ++r)
        C[(size_t)(mr + r) * N + n] = acc[mi][j][r] + bz;
    }
  }
}

// ---------------------------------------------------------------------------
// attn = sampled @ out_w^T + out_b ; src = LN(src + attn) ; sh = f16(src)
// block = 256 threads (8 waves), handles 16 rows x 256 cols
// ---------------------------------------------------------------------------
__global__ void attn_out_ln_kernel(const half_t* __restrict__ Ain, const half_t* __restrict__ W,
                                   const float* __restrict__ bias, float* __restrict__ src,
                                   const float* __restrict__ lg, const float* __restrict__ lb,
                                   half_t* __restrict__ sh) {
  __shared__ half_t As[16 * 256];
  __shared__ float  Os[16 * 256];
  __shared__ float  mu_s[16], rs_s[16];
  int tid = threadIdx.x, lane = tid & 31, wave = tid >> 5;
  int m0 = blockIdx.x * 16;
  stage_tile(Ain + (size_t)m0 * 256, As, tid);
  stage_wait();
#pragma unroll
  for (int jt = 0; jt < 2; ++jt) {
    int n0 = (wave * 2 + jt) * 16;
    v8f acc = {};
    for (int k0 = 0; k0 < 256; k0 += 32) {
      v16h a = load_a_frag(As, 256, 0, k0, lane);
      acc = wmma_f16(a, load_b_frag(W, 256, n0, k0, lane), acc);
    }
    int n  = n0 + (lane & 15);
    int mr = (lane & 16) ? 8 : 0;
    float bz = bias[n];
#pragma unroll
    for (int r = 0; r < 8; ++r) {
      int ml = mr + r;
      Os[ml * 256 + n] = acc[r] + bz + src[(size_t)(m0 + ml) * 256 + n];
    }
  }
  __syncthreads();
  if (tid < 16) {
    float s = 0.f, s2 = 0.f;
    for (int c = 0; c < 256; ++c) { float v = Os[tid * 256 + c]; s += v; s2 += v * v; }
    float m = s * (1.0f / 256.0f);
    float var = s2 * (1.0f / 256.0f) - m * m;
    mu_s[tid] = m; rs_s[tid] = rsqrtf(var + 1e-5f);
  }
  __syncthreads();
  for (int i = tid; i < 16 * 256; i += 256) {
    int ml = i >> 8, c = i & 255;
    float v = (Os[i] - mu_s[ml]) * rs_s[ml] * lg[c] + lb[c];
    src[(size_t)(m0 + ml) * 256 + c] = v;
    sh [(size_t)(m0 + ml) * 256 + c] = (half_t)v;
  }
}

// ---------------------------------------------------------------------------
// Fused FFN: h = relu(src@W1^T + b1)  (16x1024 kept in LDS, f16)
//            y = h@W2^T + b2 ; src = LN(src + y) ; sh = f16(src) ; qh = f16(src+pos)
// ---------------------------------------------------------------------------
__global__ void ffn_ln_kernel(const half_t* __restrict__ sh_in,
                              const half_t* __restrict__ W1, const float* __restrict__ b1,
                              const half_t* __restrict__ W2, const float* __restrict__ b2,
                              float* __restrict__ src, const float* __restrict__ lg,
                              const float* __restrict__ lb, const float* __restrict__ pos,
                              half_t* __restrict__ qh, half_t* __restrict__ sh_out) {
  __shared__ half_t As[16 * 256];
  __shared__ half_t Hs[16 * 1024];
  __shared__ float  Os[16 * 256];
  __shared__ float  mu_s[16], rs_s[16];
  int tid = threadIdx.x, lane = tid & 31, wave = tid >> 5;
  int m0 = blockIdx.x * 16;
  stage_tile(sh_in + (size_t)m0 * 256, As, tid);
  stage_wait();
  // GEMM1: hidden (N=1024), 8 tiles per wave
#pragma unroll
  for (int jt = 0; jt < 8; ++jt) {
    int n0 = (wave * 8 + jt) * 16;
    v8f acc = {};
    for (int k0 = 0; k0 < 256; k0 += 32) {
      v16h a = load_a_frag(As, 256, 0, k0, lane);
      acc = wmma_f16(a, load_b_frag(W1, 256, n0, k0, lane), acc);
    }
    int n  = n0 + (lane & 15);
    int mr = (lane & 16) ? 8 : 0;
    float bz = b1[n];
#pragma unroll
    for (int r = 0; r < 8; ++r) {
      float v = acc[r] + bz;
      Hs[(mr + r) * 1024 + n] = (half_t)(v > 0.f ? v : 0.f);
    }
  }
  __syncthreads();
  // GEMM2: out (N=256), K=1024
#pragma unroll
  for (int jt = 0; jt < 2; ++jt) {
    int n0 = (wave * 2 + jt) * 16;
    v8f acc = {};
    for (int k0 = 0; k0 < 1024; k0 += 32) {
      v16h a = load_a_frag(Hs, 1024, 0, k0, lane);
      acc = wmma_f16(a, load_b_frag(W2, 1024, n0, k0, lane), acc);
    }
    int n  = n0 + (lane & 15);
    int mr = (lane & 16) ? 8 : 0;
    float bz = b2[n];
#pragma unroll
    for (int r = 0; r < 8; ++r) {
      int ml = mr + r;
      Os[ml * 256 + n] = acc[r] + bz + src[(size_t)(m0 + ml) * 256 + n];
    }
  }
  __syncthreads();
  if (tid < 16) {
    float s = 0.f, s2 = 0.f;
    for (int c = 0; c < 256; ++c) { float v = Os[tid * 256 + c]; s += v; s2 += v * v; }
    float m = s * (1.0f / 256.0f);
    float var = s2 * (1.0f / 256.0f) - m * m;
    mu_s[tid] = m; rs_s[tid] = rsqrtf(var + 1e-5f);
  }
  __syncthreads();
  for (int i = tid; i < 16 * 256; i += 256) {
    int ml = i >> 8, c = i & 255;
    int m = m0 + ml;
    float v = (Os[i] - mu_s[ml]) * rs_s[ml] * lg[c] + lb[c];
    src[(size_t)m * 256 + c]    = v;
    sh_out[(size_t)m * 256 + c] = (half_t)v;
    int s = m % S_;
    qh[(size_t)m * 256 + c] = (half_t)(v + pos[(size_t)s * 256 + c]);
  }
}

// ---------------------------------------------------------------------------
// Elementwise / reduction kernels
// ---------------------------------------------------------------------------
__global__ void f32_to_f16_kernel(const float* __restrict__ in, half_t* __restrict__ out, size_t n) {
  size_t t = (size_t)blockIdx.x * 256 + threadIdx.x;
  if (t < n) out[t] = (half_t)in[t];
}

__global__ void pos_embed_kernel(const float* __restrict__ level_embed, float* __restrict__ pos) {
  int t = blockIdx.x * 256 + threadIdx.x;
  if (t >= S_ * 256) return;
  int c = t & 255, s = t >> 8;
  int lh, lw, base, l;
  if (s < 4096)      { l = 0; lh = 64; lw = 64; base = 0; }
  else if (s < 5120) { l = 1; lh = 32; lw = 32; base = 4096; }
  else if (s < 5376) { l = 2; lh = 16; lw = 16; base = 5120; }
  else               { l = 3; lh = 8;  lw = 8;  base = 5376; }
  int si = s - base, yy = si / lw, xx = si % lw;
  const float scale = 6.283185307179586f;
  float yv = (yy + 1) / (lh + 1e-6f) * scale;
  float xv = (xx + 1) / (lw + 1e-6f) * scale;
  float coord = (c < 128) ? yv : xv;
  int i = (c < 128) ? c : (c - 128);
  float expo = (float)((i >> 1) << 1) / 128.0f;
  float dim = powf(10000.0f, expo);
  float arg = coord / dim;
  float v = (i & 1) ? cosf(arg) : sinf(arg);
  pos[t] = v + level_embed[l * 256 + c];
}

__global__ void fuse_kernel(const float* __restrict__ f0, const float* __restrict__ f1,
                            const float* __restrict__ f2, const float* __restrict__ tfw,
                            int coff, int C, int hw, half_t* __restrict__ out) {
  size_t t = (size_t)blockIdx.x * 256 + threadIdx.x;
  size_t total = (size_t)4 * C * hw;
  if (t >= total) return;
  int sp = (int)(t % hw);
  int c  = (int)((t / hw) % C);
  int b  = (int)(t / ((size_t)hw * C));
  float w0 = tfw[coff + c], w1 = tfw[1408 + coff + c], w2 = tfw[2816 + coff + c];
  float mx = fmaxf(w0, fmaxf(w1, w2));
  float e0 = expf(w0 - mx), e1 = expf(w1 - mx), e2 = expf(w2 - mx);
  float inv = 1.f / (e0 + e1 + e2);
  size_t fi = ((size_t)b * C + c) * hw + sp;
  float v = (e0 * f0[fi] + e1 * f1[fi] + e2 * f2[fi]) * inv;
  out[((size_t)b * hw + sp) * C + c] = (half_t)v;
}

__global__ void gn_stats_kernel(const float* __restrict__ tmp, int hw, float* __restrict__ stats) {
  __shared__ float ss[256], ss2[256];
  int b = blockIdx.x >> 5, g = blockIdx.x & 31;
  int n = 8 * hw;
  float s = 0.f, s2 = 0.f;
  for (int i = threadIdx.x; i < n; i += 256) {
    int j = i & 7, sp = i >> 3;
    float v = tmp[((size_t)b * hw + sp) * 256 + g * 8 + j];
    s += v; s2 += v * v;
  }
  ss[threadIdx.x] = s; ss2[threadIdx.x] = s2;
  __syncthreads();
  for (int st = 128; st > 0; st >>= 1) {
    if (threadIdx.x < st) { ss[threadIdx.x] += ss[threadIdx.x + st]; ss2[threadIdx.x] += ss2[threadIdx.x + st]; }
    __syncthreads();
  }
  if (threadIdx.x == 0) {
    float mu = ss[0] / n;
    float var = ss2[0] / n - mu * mu;
    stats[(b * 32 + g) * 2]     = mu;
    stats[(b * 32 + g) * 2 + 1] = rsqrtf(var + 1e-5f);
  }
}

__global__ void gn_apply_kernel(const float* __restrict__ tmp, const float* __restrict__ stats,
                                const float* __restrict__ gg, const float* __restrict__ gb,
                                float* __restrict__ src, int hw, int soff) {
  size_t t = (size_t)blockIdx.x * 256 + threadIdx.x;
  size_t total = (size_t)4 * hw * 256;
  if (t >= total) return;
  int c  = (int)(t & 255);
  int sp = (int)((t >> 8) % hw);
  int b  = (int)((t >> 8) / hw);
  int g  = c >> 3;
  float mu = stats[(b * 32 + g) * 2], rs = stats[(b * 32 + g) * 2 + 1];
  float v = (tmp[t] - mu) * rs * gg[c] + gb[c];
  src[((size_t)b * S_ + soff + sp) * 256 + c] = v;
}

__global__ void init_qs_kernel(const float* __restrict__ src, const float* __restrict__ pos,
                               half_t* __restrict__ qh, half_t* __restrict__ sh) {
  size_t t = (size_t)blockIdx.x * 256 + threadIdx.x;
  if (t >= (size_t)MTOT * 256) return;
  int c = (int)(t & 255);
  size_t m = t >> 8;
  int s = (int)(m % S_);
  float v = src[t];
  sh[t] = (half_t)v;
  qh[t] = (half_t)(v + pos[(size_t)s * 256 + c]);
}

__global__ void softmax16_kernel(float* __restrict__ aw, int cnt) {
  int t = blockIdx.x * 256 + threadIdx.x;
  if (t >= cnt) return;
  float* p = aw + (size_t)t * 16;
  float mx = p[0];
#pragma unroll
  for (int i = 1; i < 16; ++i) mx = fmaxf(mx, p[i]);
  float e[16], s = 0.f;
#pragma unroll
  for (int i = 0; i < 16; ++i) { e[i] = expf(p[i] - mx); s += e[i]; }
  float inv = 1.f / s;
#pragma unroll
  for (int i = 0; i < 16; ++i) p[i] = e[i] * inv;
}

// one thread per (m, h, d): 4 levels x 4 points x 4 corners gathers, coalesced over d
__global__ void ms_deform_kernel(const float* __restrict__ val, const float* __restrict__ off,
                                 const float* __restrict__ aw, half_t* __restrict__ out) {
  int t = blockIdx.x * 256 + threadIdx.x;
  if (t >= MTOT * 256) return;
  int d = t & 31;
  int h = (t >> 5) & 7;
  int m = t >> 8;
  int b = m / S_;
  int s = m - b * S_;
  float rx, ry;
  {
    int lw, lh, base;
    if (s < 4096)      { lw = 64; lh = 64; base = 0; }
    else if (s < 5120) { lw = 32; lh = 32; base = 4096; }
    else if (s < 5376) { lw = 16; lh = 16; base = 5120; }
    else               { lw = 8;  lh = 8;  base = 5376; }
    int si = s - base;
    rx = ((si % lw) + 0.5f) / lw;
    ry = ((si / lw) + 0.5f) / lh;
  }
  const float* offp = off + (size_t)m * 256 + h * 32;  // (l,p,2)
  const float* awp  = aw  + (size_t)m * 128 + h * 16;  // (l,p)
  const int lbase[4] = {0, 4096, 5120, 5376};
  const int lwid[4]  = {64, 32, 16, 8};
  float acc = 0.f;
#pragma unroll
  for (int l = 0; l < 4; ++l) {
    int w = lwid[l], hh = lwid[l];
    const float* vb = val + (((size_t)b * S_ + lbase[l]) * 8 + h) * 32 + d;
    float invw = 1.0f / w, invh = 1.0f / hh;
#pragma unroll
    for (int p = 0; p < 4; ++p) {
      float gx = (rx + offp[(l * 4 + p) * 2 + 0] * invw) * w - 0.5f;
      float gy = (ry + offp[(l * 4 + p) * 2 + 1] * invh) * hh - 0.5f;
      float x0f = floorf(gx), y0f = floorf(gy);
      float wx1 = gx - x0f, wy1 = gy - y0f;
      int x0 = (int)x0f, y0 = (int)y0f;
      float sv = 0.f;
#pragma unroll
      for (int dy = 0; dy < 2; ++dy) {
#pragma unroll
        for (int dx = 0; dx < 2; ++dx) {
          int xi = x0 + dx, yi = y0 + dy;
          if (xi >= 0 && xi < w && yi >= 0 && yi < hh) {
            float wt = (dx ? wx1 : 1.f - wx1) * (dy ? wy1 : 1.f - wy1);
            sv += wt * vb[(size_t)(yi * w + xi) * 256];
          }
        }
      }
      acc += awp[l * 4 + p] * sv;
    }
  }
  out[t] = (half_t)acc;
}

// ---------------------------------------------------------------------------
// Head: conv3x3(256->64)+relu, conv3x3(64->2), bilinear resize 64 -> 256
// ---------------------------------------------------------------------------
__global__ void conv1_kernel(const float* __restrict__ src, const float* __restrict__ w,
                             const float* __restrict__ bias, float* __restrict__ out) {
  int t = blockIdx.x * 256 + threadIdx.x;
  if (t >= 4 * 64 * 64 * 64) return;
  int x  = t & 63, y = (t >> 6) & 63, oc = (t >> 12) & 63, bb = t >> 18;
  float acc = bias[oc];
  for (int ky = 0; ky < 3; ++ky) {
    int iy = y + ky - 1; if (iy < 0 || iy >= 64) continue;
    for (int kx = 0; kx < 3; ++kx) {
      int ix = x + kx - 1; if (ix < 0 || ix >= 64) continue;
      const float* sp = src + ((size_t)bb * S_ + iy * 64 + ix) * 256;
      const float* wp = w + ((size_t)oc * 256) * 9 + ky * 3 + kx;
      float a = 0.f;
      for (int ic = 0; ic < 256; ++ic) a += sp[ic] * wp[(size_t)ic * 9];
      acc += a;
    }
  }
  out[t] = fmaxf(acc, 0.f);
}

__global__ void conv2_kernel(const float* __restrict__ in, const float* __restrict__ w,
                             const float* __restrict__ bias, float* __restrict__ out) {
  int t = blockIdx.x * 256 + threadIdx.x;
  if (t >= 4 * 2 * 64 * 64) return;
  int x = t & 63, y = (t >> 6) & 63, oc = (t >> 12) & 1, bb = t >> 13;
  float acc = bias[oc];
  for (int ky = 0; ky < 3; ++ky) {
    int iy = y + ky - 1; if (iy < 0 || iy >= 64) continue;
    for (int kx = 0; kx < 3; ++kx) {
      int ix = x + kx - 1; if (ix < 0 || ix >= 64) continue;
      for (int ic = 0; ic < 64; ++ic)
        acc += in[((size_t)(bb * 64 + ic) * 64 + iy) * 64 + ix] *
               w[((size_t)(oc * 64 + ic) * 3 + ky) * 3 + kx];
    }
  }
  out[t] = acc;
}

__global__ void resize_kernel(const float* __restrict__ pred, float* __restrict__ out) {
  int t = blockIdx.x * 256 + threadIdx.x;
  if (t >= 4 * 2 * 256 * 256) return;
  int X = t & 255, Y = (t >> 8) & 255, c = (t >> 16) & 1, bb = t >> 17;
  float fy = (Y + 0.5f) * 0.25f - 0.5f;
  float fx = (X + 0.5f) * 0.25f - 0.5f;
  int y0 = (int)floorf(fy), x0 = (int)floorf(fx);
  float wy = fy - y0, wx = fx - x0;
  int y0c = min(max(y0, 0), 63), y1c = min(max(y0 + 1, 0), 63);
  int x0c = min(max(x0, 0), 63), x1c = min(max(x0 + 1, 0), 63);
  const float* p = pred + ((size_t)bb * 2 + c) * 4096;
  float v = (1.f - wy) * ((1.f - wx) * p[y0c * 64 + x0c] + wx * p[y0c * 64 + x1c]) +
            wy         * ((1.f - wx) * p[y1c * 64 + x0c] + wx * p[y1c * 64 + x1c]);
  out[t] = v;
}

// ---------------------------------------------------------------------------
// Host-side orchestration
// ---------------------------------------------------------------------------
extern "C" void kernel_launch(void* const* d_in, const int* in_sizes, int n_in,
                              void* d_out, int out_size, void* d_ws, size_t ws_size,
                              hipStream_t stream) {
  (void)in_sizes; (void)n_in; (void)out_size; (void)ws_size;
  const float* feats[3][4];
  for (int t = 0; t < 3; ++t)
    for (int l = 0; l < 4; ++l) feats[t][l] = (const float*)d_in[t * 4 + l];
  const float* tfw = (const float*)d_in[12];
  const float *proj_w[4], *proj_bv[4], *gn_g[4], *gn_b[4];
  for (int l = 0; l < 4; ++l) {
    proj_w[l]  = (const float*)d_in[13 + l * 4 + 0];
    proj_bv[l] = (const float*)d_in[13 + l * 4 + 1];
    gn_g[l]    = (const float*)d_in[13 + l * 4 + 2];
    gn_b[l]    = (const float*)d_in[13 + l * 4 + 3];
  }
  const float* level_embed = (const float*)d_in[29];
  const float* off_b = (const float*)d_in[31];
  const float* aw_b  = (const float*)d_in[33];
  const float* val_b = (const float*)d_in[35];
  const float* out_b = (const float*)d_in[37];
  const float* ln1_g = (const float*)d_in[38];
  const float* ln1_b = (const float*)d_in[39];
  const float* ff1_b = (const float*)d_in[41];
  const float* ff2_b = (const float*)d_in[43];
  const float* ln2_g = (const float*)d_in[44];
  const float* ln2_b = (const float*)d_in[45];
  const float* c1_w  = (const float*)d_in[46];
  const float* c1_b  = (const float*)d_in[47];
  const float* c2_w  = (const float*)d_in[48];
  const float* c2_b  = (const float*)d_in[49];

  // workspace carve (~127 MB total)
  char* wp_ = (char*)d_ws;
  auto carve = [&](size_t bytes) -> void* {
    void* p = (void*)wp_; wp_ += (bytes + 255) & ~(size_t)255; return p;
  };
  float*  src   = (float*)carve((size_t)MTOT * 256 * 4);
  float*  pos   = (float*)carve((size_t)S_ * 256 * 4);
  half_t* qh    = (half_t*)carve((size_t)MTOT * 256 * 2);
  half_t* sh    = (half_t*)carve((size_t)MTOT * 256 * 2);
  float*  offb  = (float*)carve((size_t)MTOT * 256 * 4);   // also proj tmp
  float*  awb   = (float*)carve((size_t)MTOT * 128 * 4);
  float*  valb  = (float*)carve((size_t)MTOT * 256 * 4);   // also conv1 output
  half_t* samph = (half_t*)carve((size_t)MTOT * 256 * 2);  // also fused f16
  half_t* offw  = (half_t*)carve((size_t)393216 * 2);
  half_t* aww   = (half_t*)carve((size_t)196608 * 2);
  half_t* valw  = (half_t*)carve((size_t)393216 * 2);
  half_t* outw  = (half_t*)carve((size_t)393216 * 2);
  half_t* ff1w  = (half_t*)carve((size_t)1572864 * 2);
  half_t* ff2w  = (half_t*)carve((size_t)1572864 * 2);
  half_t* pwh   = (half_t*)carve((size_t)360448 * 2);
  float*  stats = (float*)carve((size_t)4 * 32 * 2 * 4);
  float*  pred  = (float*)carve((size_t)4 * 2 * 64 * 64 * 4);

  auto cvt = [&](const float* in, half_t* out, size_t n) {
    f32_to_f16_kernel<<<dim3((unsigned)((n + 255) / 256)), dim3(256), 0, stream>>>(in, out, n);
  };
  // weight conversion to f16 (L2-resident, reused across many GEMM tiles)
  cvt((const float*)d_in[30], offw, 393216);
  cvt((const float*)d_in[32], aww,  196608);
  cvt((const float*)d_in[34], valw, 393216);
  cvt((const float*)d_in[36], outw, 393216);
  cvt((const float*)d_in[40], ff1w, 1572864);
  cvt((const float*)d_in[42], ff2w, 1572864);
  const size_t poff[4] = {0, 32768, 98304, 229376};
  const int    CHSa[4] = {128, 256, 512, 512};
  for (int l = 0; l < 4; ++l) cvt(proj_w[l], pwh + poff[l], (size_t)256 * CHSa[l]);

  pos_embed_kernel<<<dim3(S_ * 256 / 256), dim3(256), 0, stream>>>(level_embed, pos);

  // temporal fusion + 1x1 conv (WMMA) + GroupNorm, per level
  const int hwAr[4]   = {4096, 1024, 256, 64};
  const int soffAr[4] = {0, 4096, 5120, 5376};
  const int coffAr[4] = {0, 128, 256, 512};  // matches reference's priors
  for (int l = 0; l < 4; ++l) {
    int hw = hwAr[l], C = CHSa[l];
    size_t nfuse = (size_t)4 * C * hw;
    fuse_kernel<<<dim3((unsigned)((nfuse + 255) / 256)), dim3(256), 0, stream>>>(
        feats[0][l], feats[1][l], feats[2][l], tfw, coffAr[l], C, hw, samph);
    wmma_gemm_bias<<<dim3((4 * hw) / 32, 256 / 64), dim3(32), 0, stream>>>(
        samph, pwh + poff[l], proj_bv[l], offb, 256, C);
    gn_stats_kernel<<<dim3(128), dim3(256), 0, stream>>>(offb, hw, stats);
    size_t nap = (size_t)4 * hw * 256;
    gn_apply_kernel<<<dim3((unsigned)((nap + 255) / 256)), dim3(256), 0, stream>>>(
        offb, stats, gn_g[l], gn_b[l], src, hw, soffAr[l]);
  }
  init_qs_kernel<<<dim3(MTOT * 256 / 256), dim3(256), 0, stream>>>(src, pos, qh, sh);

  // 6 encoder layers
  for (int i = 0; i < 6; ++i) {
    const half_t* offw_i = offw + (size_t)i * 65536;
    const half_t* aww_i  = aww  + (size_t)i * 32768;
    const half_t* valw_i = valw + (size_t)i * 65536;
    const half_t* outw_i = outw + (size_t)i * 65536;
    const half_t* ff1w_i = ff1w + (size_t)i * 262144;
    const half_t* ff2w_i = ff2w + (size_t)i * 262144;

    wmma_gemm_bias<<<dim3(MTOT / 32, 4), dim3(32), 0, stream>>>(qh, offw_i, off_b + i * 256, offb, 256, 256);
    wmma_gemm_bias<<<dim3(MTOT / 32, 2), dim3(32), 0, stream>>>(qh, aww_i, aw_b + i * 128, awb, 128, 256);
    softmax16_kernel<<<dim3((MTOT * 8 + 255) / 256), dim3(256), 0, stream>>>(awb, MTOT * 8);
    wmma_gemm_bias<<<dim3(MTOT / 32, 4), dim3(32), 0, stream>>>(sh, valw_i, val_b + i * 256, valb, 256, 256);
    ms_deform_kernel<<<dim3(MTOT * 256 / 256), dim3(256), 0, stream>>>(valb, offb, awb, samph);
    attn_out_ln_kernel<<<dim3(MTOT / 16), dim3(256), 0, stream>>>(
        samph, outw_i, out_b + i * 256, src, ln1_g + i * 256, ln1_b + i * 256, sh);
    ffn_ln_kernel<<<dim3(MTOT / 16), dim3(256), 0, stream>>>(
        sh, ff1w_i, ff1_b + i * 1024, ff2w_i, ff2_b + i * 256, src,
        ln2_g + i * 256, ln2_b + i * 256, pos, qh, sh);
  }

  // segmentation head
  conv1_kernel<<<dim3(4 * 64 * 64 * 64 / 256), dim3(256), 0, stream>>>(src, c1_w, c1_b, valb);
  conv2_kernel<<<dim3((4 * 2 * 64 * 64 + 255) / 256), dim3(256), 0, stream>>>(valb, c2_w, c2_b, pred);
  resize_kernel<<<dim3(4 * 2 * 256 * 256 / 256), dim3(256), 0, stream>>>(pred, (float*)d_out);
}
